// SelfAttention_6098853560991
// MI455X (gfx1250) — matrix operand order repaired and checked by reading, not compiled, not run on hardware
//
#include <hip/hip_runtime.h>
#include <hip/hip_bf16.h>

// ---------------------------------------------------------------------------
// Self-attention, B=4, S=2048, D=1024 on gfx1250 (CDNA5, wave32).
// All four GEMMs via v_wmma_f32_16x16x32_bf16 with LDS tiling and
// double-buffered async global->LDS staging (ASYNCcnt pipeline).
// ---------------------------------------------------------------------------

typedef __bf16 bf16;
typedef __attribute__((ext_vector_type(16))) __bf16 v16bf;
typedef __attribute__((ext_vector_type(8)))  float  v8f;
typedef long long i64;

// Pointer types matching __builtin_amdgcn_global_load_async_to_lds_b128's
// signature: (v4i addrspace(1)*, v4i addrspace(3)*, imm offset, imm cpol).
typedef int v4i_ __attribute__((__vector_size__(4 * sizeof(int))));
typedef __attribute__((address_space(1))) v4i_* gp128_t;
typedef __attribute__((address_space(3))) v4i_* lp128_t;

#define B_  4
#define S_  2048
#define D_  1024

#define BT_M 128          // block tile M
#define BT_N 128          // block tile N
#define BT_K 32           // K step (one WMMA depth)
#define LDT  40           // LDS row stride in elements (32 + 8 pad, keeps 16B align)

#if defined(__has_builtin)
#  if __has_builtin(__builtin_amdgcn_global_load_async_to_lds_b128)
#    define HAVE_ASYNC 1
#  endif
#endif
#ifndef HAVE_ASYNC
#  define HAVE_ASYNC 0
#endif

#if HAVE_ASYNC
#  if __has_builtin(__builtin_amdgcn_s_wait_asynccnt)
#    define WAIT_ASYNC(n) __builtin_amdgcn_s_wait_asynccnt(n)
#  else
#    define WAIT_ASYNC(n) asm volatile("s_wait_asynccnt %0" ::"i"(n))
#  endif
#else
#  define WAIT_ASYNC(n) ((void)0)
#endif

// 16-byte global -> LDS copy (async DMA when available).
__device__ __forceinline__ void async_cp16(const bf16* g, bf16* l) {
#if HAVE_ASYNC
  __builtin_amdgcn_global_load_async_to_lds_b128(
      (gp128_t)(v4i_*)g, (lp128_t)(v4i_*)l, 0, 0);
#else
  *(v4i_*)l = *(const v4i_*)g;
#endif
}

// ---------------- f32 -> bf16 conversion ----------------
__global__ void __launch_bounds__(256)
cvt_f32_to_bf16(const float* __restrict__ in, bf16* __restrict__ out, int n) {
  int i = blockIdx.x * 256 + threadIdx.x;
  if (i < n) out[i] = (bf16)in[i];
}

// ---------------- WMMA fragment loaders (ISA 7.12.2 layouts) ----------------
// A 16x32 bf16 from row-major src (ld elements). Lane l<16: row l,
// K in {0..7,16..23}; lane l+16: K in {8..15,24..31}. -> two b128 per lane.
__device__ __forceinline__ v16bf load_a_frag(const bf16* p, int ld, int lane) {
  const bf16* q = p + (size_t)(lane & 15) * ld + ((lane >> 4) << 3);
  v16bf a;
#pragma unroll
  for (int i = 0; i < 16; ++i)
    a[i] = q[((i >> 3) << 4) + (i & 7)];
  return a;
}

// B 32x16 where B[k][n] = src[n*ld + k] (NT). Lane l: column l&15;
// lanes 0-15 hold K in {0..7,16..23}, lanes 16-31 hold K in {8..15,24..31}.
__device__ __forceinline__ v16bf load_bT_frag(const bf16* p, int ld, int lane) {
  const bf16* q = p + (size_t)(lane & 15) * ld + ((lane >> 4) << 3);
  v16bf b;
#pragma unroll
  for (int i = 0; i < 16; ++i) b[i] = q[((i >> 3) << 4) + (i & 7)];
  return b;
}

// ---------------- LDS-tiled double-buffered NT GEMM ----------------
// C = scale * (A x Bsrc^T). A: [M,K] ld=lda. Bsrc: [N,K] ld=ldb.
// Block: 256 threads (8 waves), tile 128x128; wave tile 64x32.
// F32OUT: write float C; else bf16. CTRANS: store C[n*ldc + m] (transposed).
template <bool F32OUT, bool CTRANS>
__global__ void __launch_bounds__(256)
gemm_nt(const bf16* __restrict__ A, const bf16* __restrict__ Bsrc, void* Cv,
        int M, int N, int K, int lda, int ldb, int ldc,
        i64 sA, i64 sB, i64 sC, float scale) {
  __shared__ alignas(16) bf16 As[2][BT_M * LDT];
  __shared__ alignas(16) bf16 Bs[2][BT_N * LDT];

  const int t    = threadIdx.x;
  const int lane = t & 31;
  const int wave = t >> 5;
  A    += (i64)blockIdx.z * sA;
  Bsrc += (i64)blockIdx.z * sB;
  const int m0 = blockIdx.x * BT_M;
  const int n0 = blockIdx.y * BT_N;

  // Staging: thread t copies half a row (16 elems = 32B) of A tile and B tile.
  const int srow  = t >> 1;          // 0..127
  const int shalf = (t & 1) * 16;    // 0 or 16
  const bf16* gA = A    + (size_t)(m0 + srow) * lda + shalf;
  const bf16* gB = Bsrc + (size_t)(n0 + srow) * ldb + shalf;
  const int lidx = srow * LDT + shalf;

  // Prefetch tile 0 into buffer 0.
  async_cp16(gA, &As[0][lidx]);
  async_cp16(gA + 8, &As[0][lidx + 8]);
  async_cp16(gB, &Bs[0][lidx]);
  async_cp16(gB + 8, &Bs[0][lidx + 8]);

  const int mw = (wave >> 2) * 64;   // wave M offset: 0 or 64
  const int nw = (wave & 3) * 32;    // wave N offset: 0,32,64,96
  v8f acc[4][2] = {};

  const int KT = K / BT_K;
  for (int kt = 0; kt < KT; ++kt) {
    const int cur = kt & 1;
    if (kt + 1 < KT) {               // stage next tile into the other buffer
      const bf16* nA = gA + (size_t)(kt + 1) * BT_K;
      const bf16* nB = gB + (size_t)(kt + 1) * BT_K;
      bf16* lA = &As[cur ^ 1][lidx];
      bf16* lB = &Bs[cur ^ 1][lidx];
      async_cp16(nA, lA);
      async_cp16(nA + 8, lA + 8);
      async_cp16(nB, lB);
      async_cp16(nB + 8, lB + 8);
      WAIT_ASYNC(4);                 // the 4 older copies (current tile) done
    } else {
      WAIT_ASYNC(0);
    }
    __syncthreads();                 // current tile visible to all waves

    v16bf af[4], bfr[2];
#pragma unroll
    for (int mi = 0; mi < 4; ++mi)
      af[mi] = load_a_frag(&As[cur][(mw + mi * 16) * LDT], LDT, lane);
#pragma unroll
    for (int nj = 0; nj < 2; ++nj)
      bfr[nj] = load_bT_frag(&Bs[cur][(nw + nj * 16) * LDT], LDT, lane);

#pragma unroll
    for (int mi = 0; mi < 4; ++mi)
#pragma unroll
      for (int nj = 0; nj < 2; ++nj)
        acc[mi][nj] = __builtin_amdgcn_wmma_f32_16x16x32_bf16(
            false, af[mi], false, bfr[nj], (short)0, acc[mi][nj], false, false);

    __syncthreads();                 // all reads done before buffer is rewritten
  }

  // C/D layout: VGPR r -> row r (lanes 0-15) / row 8+r (lanes 16-31).
  const int col  = lane & 15;
  const int rsel = (lane >> 4) * 8;
  float* fC = (float*)Cv + (CTRANS ? 0 : (i64)blockIdx.z * sC);
  bf16*  bC = (bf16*)Cv + (CTRANS ? 0 : (i64)blockIdx.z * sC);
#pragma unroll
  for (int mi = 0; mi < 4; ++mi) {
#pragma unroll
    for (int nj = 0; nj < 2; ++nj) {
#pragma unroll
      for (int r = 0; r < 8; ++r) {
        const int row = m0 + mw + mi * 16 + rsel + r;
        const int cc  = n0 + nw + nj * 16 + col;
        const size_t idx = CTRANS ? (size_t)cc * ldc + row
                                  : (size_t)row * ldc + cc;
        const float v = acc[mi][nj][r] * scale;
        if (F32OUT) fC[idx] = v;
        else        bC[idx] = (bf16)v;
      }
    }
  }
}

// ---------------- row softmax over bf16 scores, in place ----------------
__global__ void __launch_bounds__(256)
softmax_rows(bf16* __restrict__ sc) {
  __shared__ float red[256];
  bf16* p = sc + (size_t)blockIdx.x * S_;
  const int t = threadIdx.x;

  float e[S_ / 256];
  float lmax = -3.4e38f;
#pragma unroll
  for (int j = 0; j < S_ / 256; ++j) {
    float v = (float)p[t + j * 256];
    e[j] = v;
    lmax = fmaxf(lmax, v);
  }
  red[t] = lmax;
  __syncthreads();
  for (int s = 128; s > 0; s >>= 1) {
    if (t < s) red[t] = fmaxf(red[t], red[t + s]);
    __syncthreads();
  }
  const float rmax = red[0];
  __syncthreads();

  float lsum = 0.f;
#pragma unroll
  for (int j = 0; j < S_ / 256; ++j) {
    e[j] = __expf(e[j] - rmax);
    lsum += e[j];
  }
  red[t] = lsum;
  __syncthreads();
  for (int s = 128; s > 0; s >>= 1) {
    if (t < s) red[t] += red[t + s];
    __syncthreads();
  }
  const float inv = 1.0f / red[0];
#pragma unroll
  for (int j = 0; j < S_ / 256; ++j)
    p[t + j * 256] = (bf16)(e[j] * inv);
}

// ---------------- launch ----------------
extern "C" void kernel_launch(void* const* d_in, const int* in_sizes, int n_in,
                              void* d_out, int out_size, void* d_ws, size_t ws_size,
                              hipStream_t stream) {
  const float* x  = (const float*)d_in[0];
  const float* Wk = (const float*)d_in[1];
  const float* Wq = (const float*)d_in[2];
  const float* Wv = (const float*)d_in[3];

  const size_t nX = (size_t)B_ * S_ * D_;   // 8,388,608
  const size_t nW = (size_t)D_ * D_;        // 1,048,576
  const int    M  = B_ * S_;                // 8192

  char* ws = (char*)d_ws;
  bf16* x_bf  = (bf16*)ws;                                      // 16 MiB
  bf16* w_bf  = (bf16*)(ws + nX * 2);                           //  6 MiB (Wq,Wk,Wv)
  bf16* qk_bf = (bf16*)(ws + nX * 2 + 3 * nW * 2);              // 32 MiB (Q,K)
  bf16* vt_bf = (bf16*)(ws + nX * 2 + 3 * nW * 2 + 2 * nX * 2); // 16 MiB  V^T [D, B*S]
  bf16* sc_bf = (bf16*)(ws + nX * 2 + 3 * nW * 2 + 3 * nX * 2); // 32 MiB scores

  // 1) f32 -> bf16 (weight slots: 0=query, 1=key, 2=value)
  cvt_f32_to_bf16<<<(nX + 255) / 256, 256, 0, stream>>>(x,  x_bf,          (int)nX);
  cvt_f32_to_bf16<<<(nW + 255) / 256, 256, 0, stream>>>(Wq, w_bf + 0 * nW, (int)nW);
  cvt_f32_to_bf16<<<(nW + 255) / 256, 256, 0, stream>>>(Wk, w_bf + 1 * nW, (int)nW);
  cvt_f32_to_bf16<<<(nW + 255) / 256, 256, 0, stream>>>(Wv, w_bf + 2 * nW, (int)nW);

  // 2a) Q,K projection: y = x @ W^T. z picks weight/output slot.
  {
    dim3 grid(M / BT_M, D_ / BT_N, 2);
    gemm_nt<false, false><<<grid, 256, 0, stream>>>(
        x_bf, w_bf, qk_bf, M, D_, D_, D_, D_, D_,
        (i64)0, (i64)nW, (i64)nX, 1.0f);
  }
  // 2b) V projection stored transposed: Vt[d, b*S+s] (ldc = M).
  {
    dim3 grid(M / BT_M, D_ / BT_N, 1);
    gemm_nt<false, true><<<grid, 256, 0, stream>>>(
        x_bf, w_bf + 2 * nW, vt_bf, M, D_, D_, D_, D_, M,
        (i64)0, (i64)0, (i64)0, 1.0f);
  }

  // 3) scores = (Q @ K^T) / sqrt(D) per batch.
  {
    dim3 grid(S_ / BT_M, S_ / BT_N, B_);
    gemm_nt<false, false><<<grid, 256, 0, stream>>>(
        qk_bf /*Q*/, qk_bf + nX /*K*/, sc_bf, S_, S_, D_, D_, D_, S_,
        (i64)S_ * D_, (i64)S_ * D_, (i64)S_ * S_, 0.03125f /* 1/sqrt(1024) */);
  }

  // 4) softmax along key axis, in place.
  softmax_rows<<<B_ * S_, 256, 0, stream>>>(sc_bf);

  // 5) context = P @ V = P @ (Vt)^T per batch, f32 out. Bsrc = Vt rows (ldb = M),
  //    per-batch column offset = b*S -> strideB = S_.
  {
    dim3 grid(S_ / BT_M, D_ / BT_N, B_);
    gemm_nt<true, false><<<grid, 256, 0, stream>>>(
        sc_bf, vt_bf, (float*)d_out, S_, D_, S_, S_, M, D_,
        (i64)S_ * S_, (i64)S_, (i64)S_ * D_, 1.0f);
  }
}
